// SequentialMoEDispatch_17935783428805
// MI455X (gfx1250) — matrix operand order repaired and checked by reading, not compiled, not run on hardware
//
#include <hip/hip_runtime.h>
#include <hip/hip_bf16.h>

// Problem constants (from reference)
#define E_ 8
#define H_ 1024
#define I_ 2816
#define B_ 256
#define K_ 2
#define SLOTS_PER_E 256
#define NSLOT (E_ * SLOTS_PER_E)

typedef __attribute__((ext_vector_type(16))) __bf16 v16bf;
typedef __attribute__((ext_vector_type(8)))  __bf16 v8bf;
typedef __attribute__((ext_vector_type(2)))  __bf16 v2bf;
typedef __attribute__((ext_vector_type(8)))  float  v8f;
typedef __attribute__((ext_vector_type(2)))  float  v2f;
typedef __attribute__((ext_vector_type(4)))  unsigned int u32x4;
typedef __attribute__((ext_vector_type(8)))  int i32x8;
typedef __attribute__((ext_vector_type(4)))  int i32x4;

#if __has_builtin(__builtin_amdgcn_rcpf)
#define FAST_RCP(x) __builtin_amdgcn_rcpf(x)
#else
#define FAST_RCP(x) (1.0f / (x))
#endif

// -------- workspace layout (bytes) --------
static constexpr size_t O_CNT  = 0;
static constexpr size_t O_TIDX = 256;
static constexpr size_t O_TW   = O_TIDX + (size_t)NSLOT * 4;
static constexpr size_t O_POS  = O_TW   + (size_t)NSLOT * 4;
static constexpr size_t O_ACT  = (O_POS + (size_t)B_ * K_ * 4 + 255) & ~(size_t)255;
static constexpr size_t O_Y    = O_ACT + (size_t)NSLOT * I_ * 2;

// =============================================================
// TDM: 2D fp32 tile load (tile_h rows x tile_w cols) from global
// row-major (row pitch = row_stride elems) into LDS at lds_off.
// D# fields per CDNA5 ISA 8.3/8.4; 6-arg builtin on this toolchain.
// =============================================================
__device__ __forceinline__
void tdm_load_tile_f32(unsigned lds_off, const float* gaddr,
                       int tile_w, int tile_h, int row_stride) {
    unsigned long long ga = (unsigned long long)(size_t)gaddr;
    u32x4 g0;
    g0[0] = 1u;                                   // count=1, user descriptor
    g0[1] = lds_off;                              // lds_addr (bytes)
    g0[2] = (unsigned)(ga & 0xFFFFFFFFu);         // global_addr[31:0]
    g0[3] = (unsigned)((ga >> 32) & 0x01FFFFFFu)  // global_addr[56:32]
          | (2u << 30);                           // type=2 ("image")
    i32x8 g1;
    g1[0] = (2 << 16);                            // data_size=2 (4 bytes)
    g1[1] = (row_stride & 0xFFFF) << 16;          // tensor_dim0[15:0]
    g1[2] = ((row_stride >> 16) & 0xFFFF)         // tensor_dim0[31:16]
          | (0x4000 << 16);                       // tensor_dim1[15:0] (large)
    g1[3] = (tile_w & 0xFFFF) << 16;              // tensor_dim1 hi | tile_dim0
    g1[4] = (tile_h & 0xFFFF);                    // tile_dim1 (tile_dim2=0)
    g1[5] = row_stride;                           // tensor_dim0_stride[31:0]
    g1[6] = 0;
    g1[7] = 0;
    i32x4 z4 = {0, 0, 0, 0};                      // groups 2/3 unused (2D)
    i32x8 z8 = {0, 0, 0, 0, 0, 0, 0, 0};
    __builtin_amdgcn_tensor_load_to_lds(g0, g1, z4, z4, z8, 0);
}

// =============================================================
// Kernel 1: routing — build per-expert token lists (1 block)
// =============================================================
__global__ void moe_route(const long long* __restrict__ ids,
                          const float* __restrict__ ew,
                          int* __restrict__ cnt, int* __restrict__ tidx,
                          float* __restrict__ tw, int* __restrict__ pos) {
    const int b = threadIdx.x;            // 256 threads == B tokens
    if (b < E_) cnt[b] = 0;
    __syncthreads();
    #pragma unroll
    for (int k = 0; k < K_; ++k) {
        int e = (int)ids[b * K_ + k];
        int s = atomicAdd(&cnt[e], 1);
        int gs = e * SLOTS_PER_E + s;
        tidx[gs] = b;
        tw[gs]   = ew[b * K_ + k];
        pos[b * K_ + k] = gs;
    }
}

// =============================================================
// Kernel 2: act = silu(x@Wg) * (x@Wu) for routed (token,expert)
// grid: (E, I/128, 16 token tiles), block: 256 = 8 waves
// Weights streamed via TDM into LDS (32x128 fp32 tiles, gate+up);
// x chunk staged cooperatively as bf16; one 16x16 WMMA tile/wave.
// =============================================================
__global__ __launch_bounds__(256)
void moe_gateup(const float* __restrict__ x,
                const float* __restrict__ gw, const float* __restrict__ uw,
                const int* __restrict__ cnt, const int* __restrict__ tidx,
                __bf16* __restrict__ act) {
    const int e  = blockIdx.x;
    const int it = blockIdx.y;    // 0..21  (I/128)
    const int tt = blockIdx.z;    // 0..15  (token tiles of 16)
    const int n  = cnt[e];
    if (tt * 16 >= n) return;

    __shared__ __bf16 xsc[16][32];     // 1 KB  : x chunk (16 tok x 32 h)
    __shared__ float  wgs[32][128];    // 16 KB : gate tile (TDM dest)
    __shared__ float  wus[32][128];    // 16 KB : up   tile (TDM dest)
    __shared__ int    stok[16];

    const int tid = threadIdx.x;
    if (tid < 16) {
        int s = tt * 16 + tid;
        if (s >= n) s = n - 1;
        stok[tid] = tidx[e * SLOTS_PER_E + s];
    }
    __syncthreads();

    const int lane = tid & 31;
    const int wave = tid >> 5;
    const int m    = lane & 15;       // A row / B,C column within tile
    const int hi   = lane >> 4;
    const int i0   = it * 128;
    const int lcol = wave * 16 + m;   // local column 0..127

    // Per-thread fixed x-staging slot: row xr (0..15), cols xc..xc+1.
    const int xr = tid >> 4;          // 16 threads per row
    const int xc = (tid & 15) * 2;
    const float* xp = x + (size_t)stok[xr] * H_ + xc;  // hoisted; +kk in loop
    v2bf* xdst = (v2bf*)&xsc[xr][xc];

    const unsigned off_wg = (unsigned)(size_t)&wgs[0][0];
    const unsigned off_wu = (unsigned)(size_t)&wus[0][0];
    const float* gbase = gw + ((size_t)e * H_) * I_ + i0;
    const float* ubase = uw + ((size_t)e * H_) * I_ + i0;

    v8f accg = {}; v8f accu = {};

    for (int kk = 0; kk < H_; kk += 32) {
        __syncthreads();   // previous chunk's LDS reads complete
        if (wave == 0) {   // TDM: DMA both weight tiles into LDS
            tdm_load_tile_f32(off_wg, gbase + (size_t)kk * I_, 128, 32, I_);
            tdm_load_tile_f32(off_wu, ubase + (size_t)kk * I_, 128, 32, I_);
        }
        // cooperative x chunk stage (f32x2 -> packed bf16x2), overlaps DMA
        {
            v2f xv = *(const v2f*)(xp + kk);
            v2bf p; p[0] = (__bf16)xv[0]; p[1] = (__bf16)xv[1];
            *xdst = p;
        }
        if (wave == 0) __builtin_amdgcn_s_wait_tensorcnt(0);
        __syncthreads();

        // A fragment (16x32 bf16) from LDS chunk
        v8bf alo = *(const v8bf*)&xsc[m][hi * 8];
        v8bf ahi = *(const v8bf*)&xsc[m][16 + hi * 8];
        v16bf a;
        #pragma unroll
        for (int j = 0; j < 8; ++j) { a[j] = alo[j]; a[8 + j] = ahi[j]; }

        // B fragments from TDM-staged LDS tiles (conflict-free reads)
        v16bf bg, bu;
        #pragma unroll
        for (int j = 0; j < 16; ++j) {
            bg[j] = (__bf16)wgs[hi * 8 + j + (j >= 8 ? 8 : 0)][lcol];
            bu[j] = (__bf16)wus[hi * 8 + j + (j >= 8 ? 8 : 0)][lcol];
        }
        accg = __builtin_amdgcn_wmma_f32_16x16x32_bf16(false, a, false, bg,
                                                       (short)0, accg, false, false);
        accu = __builtin_amdgcn_wmma_f32_16x16x32_bf16(false, a, false, bu,
                                                       (short)0, accu, false, false);
    }

    // SiLU(g) * u -> act (bf16); fast sigmoid via v_exp + v_rcp
    const int slot0 = e * SLOTS_PER_E + tt * 16;
    #pragma unroll
    for (int r = 0; r < 8; ++r) {
        int   row = hi * 8 + r;
        float g   = accg[r];
        float v   = g * FAST_RCP(1.0f + __expf(-g)) * accu[r];
        act[(size_t)(slot0 + row) * I_ + i0 + lcol] = (__bf16)v;
    }
}

// =============================================================
// Kernel 3: y[slot] = tokW[slot] * (act[slot] @ Wd[e])
// grid: (E, H/128, 16 slot tiles), block: 256 = 8 waves
// Down-weights streamed via TDM; A (bf16 act) loaded direct from global.
// =============================================================
__global__ __launch_bounds__(256)
void moe_down(const __bf16* __restrict__ act, const float* __restrict__ dw,
              const int* __restrict__ cnt, const float* __restrict__ tw,
              float* __restrict__ y) {
    const int e  = blockIdx.x;
    const int ht = blockIdx.y;    // 0..7 (H/128)
    const int st = blockIdx.z;    // 0..15
    const int n  = cnt[e];
    if (st * 16 >= n) return;

    __shared__ float wds[32][128]; // 16 KB : down tile (TDM dest)
    __shared__ float sw[16];

    const int tid = threadIdx.x;
    if (tid < 16) {
        int s = st * 16 + tid;
        sw[tid] = (s < n) ? tw[e * SLOTS_PER_E + s] : 0.0f;
    }
    __syncthreads();

    const int lane = tid & 31;
    const int wave = tid >> 5;
    const int m    = lane & 15;
    const int hi   = lane >> 4;
    const int h0   = ht * 128;
    const int lcol = wave * 16 + m;
    const int slot0 = e * SLOTS_PER_E + st * 16;

    const unsigned off_wd = (unsigned)(size_t)&wds[0][0];
    const __bf16* arow = act + (size_t)(slot0 + m) * I_;   // hoisted
    const float*  dbase = dw + ((size_t)e * I_) * H_ + h0;

    v8f acc = {};
    for (int kk = 0; kk < I_; kk += 32) {
        __syncthreads();
        if (wave == 0) {
            tdm_load_tile_f32(off_wd, dbase + (size_t)kk * H_, 128, 32, H_);
        }
        // A fragment: two contiguous 16B bf16 loads per lane (overlaps DMA)
        v8bf alo = *(const v8bf*)&arow[kk + hi * 8];
        v8bf ahi = *(const v8bf*)&arow[kk + 16 + hi * 8];
        if (wave == 0) __builtin_amdgcn_s_wait_tensorcnt(0);
        __syncthreads();

        v16bf a;
        #pragma unroll
        for (int j = 0; j < 8; ++j) { a[j] = alo[j]; a[8 + j] = ahi[j]; }
        v16bf b;
        #pragma unroll
        for (int j = 0; j < 16; ++j)
            b[j] = (__bf16)wds[hi * 8 + j + (j >= 8 ? 8 : 0)][lcol];

        acc = __builtin_amdgcn_wmma_f32_16x16x32_bf16(false, a, false, b,
                                                      (short)0, acc, false, false);
    }

    #pragma unroll
    for (int r = 0; r < 8; ++r) {
        int row = hi * 8 + r;
        y[(size_t)(slot0 + row) * H_ + h0 + lcol] = acc[r] * sw[row];
    }
}

// =============================================================
// Kernel 4: out[b,h] = y[pos[b,0],h] + y[pos[b,1],h]  (deterministic)
// =============================================================
__global__ void moe_combine(const float* __restrict__ y,
                            const int* __restrict__ pos,
                            float* __restrict__ out) {
    int idx = blockIdx.x * blockDim.x + threadIdx.x;
    if (idx >= B_ * H_) return;
    int b = idx >> 10;
    int h = idx & (H_ - 1);
    out[idx] = y[(size_t)pos[b * K_ + 0] * H_ + h] +
               y[(size_t)pos[b * K_ + 1] * H_ + h];
}

extern "C" void kernel_launch(void* const* d_in, const int* in_sizes, int n_in,
                              void* d_out, int out_size, void* d_ws, size_t ws_size,
                              hipStream_t stream) {
    const float*     x   = (const float*)d_in[0];
    const long long* ids = (const long long*)d_in[1];
    const float*     ew  = (const float*)d_in[2];
    const float*     gw  = (const float*)d_in[3];
    const float*     uw  = (const float*)d_in[4];
    const float*     dw  = (const float*)d_in[5];
    float*           out = (float*)d_out;

    char* ws = (char*)d_ws;
    int*    cnt  = (int*)   (ws + O_CNT);
    int*    tidx = (int*)   (ws + O_TIDX);
    float*  tw   = (float*) (ws + O_TW);
    int*    pos  = (int*)   (ws + O_POS);
    __bf16* act  = (__bf16*)(ws + O_ACT);
    float*  y    = (float*) (ws + O_Y);

    moe_route<<<1, 256, 0, stream>>>(ids, ew, cnt, tidx, tw, pos);
    moe_gateup<<<dim3(E_, I_ / 128, 16), 256, 0, stream>>>(x, gw, uw, cnt, tidx, act);
    moe_down<<<dim3(E_, H_ / 128, 16), 256, 0, stream>>>(act, dw, cnt, tw, y);
    moe_combine<<<(B_ * H_ + 255) / 256, 256, 0, stream>>>(y, pos, out);
}
